// EquivariantMolecularEnergy_83270825935551
// MI455X (gfx1250) — matrix operand order repaired and checked by reading, not compiled, not run on hardware
//
#include <hip/hip_runtime.h>
#include <hip/hip_bf16.h>
#include <math.h>

// ---------------- Problem constants ----------------
#define HID      128
#define NNODES   10000
#define NEDGES   160000
#define NLAYERS  6
#define NRBF     50
#define NT       8            // 8 N-tiles of 16 -> 128 output cols
#define KP1      320          // mw1 K padded 307 -> 320
#define EF_STRIDE 164         // padded LDS row stride (uints) to break bank conflicts
#define H_STRIDE  68          // padded LDS row stride (uints) for 128-wide tiles

// ---------------- Types ----------------
typedef __attribute__((ext_vector_type(16))) __bf16 v16bf;
typedef __attribute__((ext_vector_type(8)))  float  v8f;

union Frag { uint4 q[2]; v16bf v; };

// ---------------- Helpers ----------------
__device__ __forceinline__ unsigned short f2bf(float f) {
    unsigned u = __float_as_uint(f);
    unsigned r = u + 0x7FFFu + ((u >> 16) & 1u);   // round-to-nearest-even
    return (unsigned short)(r >> 16);
}
__device__ __forceinline__ unsigned packbf(float a, float b) {
    return (unsigned)f2bf(a) | ((unsigned)f2bf(b) << 16);
}
__device__ __forceinline__ float bf2f(unsigned short h) {
    return __uint_as_float(((unsigned)h) << 16);
}
__device__ __forceinline__ float silu_f(float x) {
    return x / (1.f + __expf(-x));
}

// CDNA5 async global->LDS copy (ASYNCcnt-tracked, bypasses VGPRs).
// lds_off: wave-relative LDS byte address (low 32 bits of a flat LDS pointer).
__device__ __forceinline__ void async_g2l_b128(unsigned lds_off, const void* gaddr) {
    asm volatile("global_load_async_to_lds_b128 %0, %1, off"
                 :: "v"(lds_off), "v"(gaddr) : "memory");
}
__device__ __forceinline__ void wait_async0() {
    asm volatile("s_wait_asynccnt 0x0" ::: "memory");
}
__device__ __forceinline__ unsigned lds_off_of(const void* p) {
    return (unsigned)(uintptr_t)p;   // flat LDS addr low 32 bits == DS offset
}

__device__ __forceinline__ v8f wmma_bf(const Frag& a, const Frag& b, v8f c) {
    return __builtin_amdgcn_wmma_f32_16x16x32_bf16(
        /*neg_a=*/false, a.v, /*neg_b=*/false, b.v,
        /*c_mod=*/(short)0, c, /*reuse_a=*/false, /*reuse_b=*/false);
}

// One 16xK @ Kx128 tile: A rows from LDS (bf16 pairs, `stride` uints/row),
// B from fragment-packed weights, bias-initialized accumulators.
__device__ __forceinline__ void gemm_tile(const unsigned* lds_base, int stride,
                                          int ksteps, const unsigned* __restrict__ wp,
                                          const float* __restrict__ bias,
                                          int mbase, int half, int ln, int lane,
                                          v8f* acc) {
    #pragma unroll
    for (int nt = 0; nt < NT; ++nt) {
        float b = bias[nt * 16 + ln];
        #pragma unroll
        for (int i = 0; i < 8; ++i) acc[nt][i] = b;
    }
    const unsigned* arow = lds_base + (mbase + ln) * stride;
    for (int ks = 0; ks < ksteps; ++ks) {
        Frag a;
        a.q[0] = *(const uint4*)(arow + ks * 16 + half * 8);       // VGPR j=0..3
        a.q[1] = *(const uint4*)(arow + ks * 16 + half * 8 + 4);   // VGPR j=4..7
        #pragma unroll
        for (int nt = 0; nt < NT; ++nt) {
            Frag b;
            const uint4* bp = (const uint4*)&wp[((ks * NT + nt) * 32 + lane) * 8];
            b.q[0] = bp[0];
            b.q[1] = bp[1];
            acc[nt] = wmma_bf(a, b, acc[nt]);
        }
    }
}

// Store D fragments (optionally through silu) as bf16 into an LDS tile.
__device__ __forceinline__ void store_bf_tile(unsigned short* dst, int stride_bf,
                                              int mbase, int half, int ln,
                                              const v8f* acc, bool dosilu) {
    #pragma unroll
    for (int v = 0; v < 8; ++v) {
        int row = mbase + half * 8 + v;
        #pragma unroll
        for (int nt = 0; nt < NT; ++nt) {
            float x = acc[nt][v];
            if (dosilu) x = silu_f(x);
            dst[row * stride_bf + nt * 16 + ln] = f2bf(x);
        }
    }
}

// ---------------- Prep kernels ----------------

// feats0 = bf16(embed[atomic_numbers]); copy coords
__global__ __launch_bounds__(256) void init_kernel(const int* __restrict__ an,
                                                   const float* __restrict__ embed,
                                                   const float* __restrict__ coords,
                                                   unsigned* __restrict__ feats_pairs,
                                                   float* __restrict__ coordsA) {
    int idx = blockIdx.x * blockDim.x + threadIdx.x;
    if (idx < NNODES * 64) {
        int n = idx >> 6, p = idx & 63;
        const float* e = &embed[an[n] * HID];
        feats_pairs[idx] = packbf(e[2 * p], e[2 * p + 1]);
    }
    if (idx < NNODES * 3) coordsA[idx] = coords[idx];
}

// Static RBF*cutoff tail for each edge: 25 bf16 pairs (+7 zero pairs).
__global__ __launch_bounds__(256) void edge_tail_kernel(const int* __restrict__ erow,
                                                        const int* __restrict__ ecol,
                                                        const float* __restrict__ coords,
                                                        unsigned* __restrict__ tail) {
    int e = blockIdx.x * blockDim.x + threadIdx.x;
    if (e >= NEDGES) return;
    int r = erow[e], c = ecol[e];
    float dx = coords[r * 3 + 0] - coords[c * 3 + 0];
    float dy = coords[r * 3 + 1] - coords[c * 3 + 1];
    float dz = coords[r * 3 + 2] - coords[c * 3 + 2];
    float d = sqrtf(dx * dx + dy * dy + dz * dz);
    float cut = (d < 10.f) ? 0.5f * (__cosf(3.14159265358979f * d * 0.1f) + 1.f) : 0.f;
    float a[NRBF];
    const float winv = 1.f / 0.2f;   // width = CUTOFF/N_RBF
    #pragma unroll
    for (int i = 0; i < NRBF; ++i) {
        float ci = (10.f / 49.f) * (float)i;   // linspace(0,10,50)
        float dd = d - ci;
        a[i] = __expf(-dd * dd * winv) * cut;
    }
    unsigned* t = &tail[e * 32];
    #pragma unroll
    for (int i = 0; i < 25; ++i) t[i] = packbf(a[2 * i], a[2 * i + 1]);
    #pragma unroll
    for (int i = 25; i < 32; ++i) t[i] = 0u;
}

// Repack weights [K x 128] f32 into bf16 B-fragment layout:
// dst[((ks*8+nt)*32+lane)*8 + j] = pair(W[k0+2j][n], W[k0+2j+1][n]),
// k0 = ks*32 + (lane>=16 ? 16 : 0), n = nt*16 + lane%16.
// permute_mw1: map padded K -> [feats_r(0..127), feats_c(128..255), rbf(256..305),
//                               edist(306), zero-pad(307..319)] from the reference's
//              [fr, fc, edist(256), rbf(257..306)] row order.
__global__ __launch_bounds__(256) void pack_kernel(const float* __restrict__ W,
                                                   unsigned* __restrict__ dst,
                                                   int ksteps, int Ksrc, int permute_mw1) {
    int id = blockIdx.x * blockDim.x + threadIdx.x;
    if (id >= ksteps * NT * 32 * 8) return;
    int j = id & 7;
    int lane = (id >> 3) & 31;
    int nt = (id >> 8) & 7;
    int ks = id >> 11;
    int n = nt * 16 + (lane & 15);
    int k0 = ks * 32 + ((lane & 16) ? 16 : 0) + 2 * j;
    float v01[2];
    #pragma unroll
    for (int s = 0; s < 2; ++s) {
        int k = k0 + s;
        int src;
        if (permute_mw1) {
            if (k < 256)       src = k;
            else if (k < 306)  src = 257 + (k - 256);
            else if (k == 306) src = 256;
            else               src = -1;
        } else {
            src = (k < Ksrc) ? k : -1;
        }
        v01[s] = (src >= 0) ? W[src * HID + n] : 0.f;
    }
    dst[id] = packbf(v01[0], v01[1]);
}

// Zero agg; coords_out = coords_in (double-buffer for functional coord update)
__global__ __launch_bounds__(256) void layer_prep_kernel(float* __restrict__ agg,
                                                         const float* __restrict__ cin,
                                                         float* __restrict__ cout) {
    int idx = blockIdx.x * blockDim.x + threadIdx.x;
    if (idx < NNODES * HID) agg[idx] = 0.f;
    if (idx < NNODES * 3) cout[idx] = cin[idx];
}

__global__ void zero_out_kernel(float* out) { if (threadIdx.x == 0) out[0] = 0.f; }

// ---------------- Fused per-layer edge kernel ----------------
// 64 edges/block, 4 waves, each wave a 16-edge M-tile.
// ef(LDS via async-to-LDS) -> silu(GEMM1) -> silu(GEMM2)=m -> agg atomics
//                          -> silu(GEMM3) -> dot(cw2) = w -> coord atomics.
__global__ __launch_bounds__(128) void edge_kernel(
    const unsigned* __restrict__ feats_pairs,     // [N][64] bf16 pairs
    const float* __restrict__ coords_in,
    float* __restrict__ coords_out,
    float* __restrict__ agg,                      // [N][128] f32
    const int* __restrict__ erow, const int* __restrict__ ecol,
    const unsigned* __restrict__ tail,            // [E][32] bf16 pairs
    const unsigned* __restrict__ w1p, const float* __restrict__ b1,
    const unsigned* __restrict__ w2p, const float* __restrict__ b2,
    const unsigned* __restrict__ c1p, const float* __restrict__ cb1,
    const float* __restrict__ cw2, const float* __restrict__ cb2) {
    __shared__ unsigned ef[64 * EF_STRIDE];   // 41984 B  (ef tile, later h3 tile)
    __shared__ unsigned hb[64 * H_STRIDE];    // 17408 B  (h1 tile, later m tile)
    __shared__ float    dif[64 * 3];
    __shared__ int      rown[64];

    const int tid  = threadIdx.x;
    const int lane = tid & 31;
    const int wave = tid >> 5;
    const int half = (lane >> 4) & 1;
    const int ln   = lane & 15;
    const int e0   = blockIdx.x * 64;
    const int mbase = wave * 16;

    // ---- phase 0+1 fused: kick off async ef gathers, then do geometry ----
    {
        int e  = tid >> 1;
        int ge = e0 + e;
        int r  = erow[ge];
        int c  = ecol[ge];
        unsigned lrow = lds_off_of(&ef[e * EF_STRIDE]);
        if ((tid & 1) == 0) {
            // K 0..127 (feats of row node) + K 128..159 (first 1/4 of col feats)
            const uint4* fr = (const uint4*)&feats_pairs[r * 64];
            #pragma unroll
            for (int i = 0; i < 16; ++i) async_g2l_b128(lrow + i * 16, fr + i);
            const uint4* fc = (const uint4*)&feats_pairs[c * 64];
            #pragma unroll
            for (int i = 0; i < 4; ++i) async_g2l_b128(lrow + 256 + i * 16, fc + i);
        } else {
            // K 160..255 (rest of col feats) + rbf tail K 256..303
            const uint4* fc = (const uint4*)&feats_pairs[c * 64];
            #pragma unroll
            for (int i = 4; i < 16; ++i) async_g2l_b128(lrow + 256 + i * 16, fc + i);
            const uint4* tp = (const uint4*)&tail[ge * 32];
            #pragma unroll
            for (int i = 0; i < 6; ++i) async_g2l_b128(lrow + 512 + i * 16, tp + i);
        }
        if (tid < 64) {
            // per-edge geometry overlaps with the async DMA above
            int ee = e0 + tid;
            int rr = erow[ee], cc = ecol[ee];
            rown[tid] = rr;
            float dx = coords_in[rr * 3 + 0] - coords_in[cc * 3 + 0];
            float dy = coords_in[rr * 3 + 1] - coords_in[cc * 3 + 1];
            float dz = coords_in[rr * 3 + 2] - coords_in[cc * 3 + 2];
            dif[tid * 3 + 0] = dx;
            dif[tid * 3 + 1] = dy;
            dif[tid * 3 + 2] = dz;
            unsigned* row = &ef[tid * EF_STRIDE];
            row[152] = tail[ee * 32 + 24];                        // rbf K 304..305
            row[153] = packbf(dx * dx + dy * dy + dz * dz, 0.f);  // edist @ K=306
            row[154] = 0u; row[155] = 0u;                         // K 308..311
            *(uint4*)(row + 156) = make_uint4(0u, 0u, 0u, 0u);    // K 312..319
        }
        wait_async0();          // own wave's async copies landed in LDS
    }
    __syncthreads();            // all waves' copies visible

    v8f acc[NT];
    // ---- GEMM1: h1 = silu(ef @ mw1 + b1), K = 320 ----
    gemm_tile(ef, EF_STRIDE, KP1 / 32, w1p, b1, mbase, half, ln, lane, acc);
    store_bf_tile((unsigned short*)hb, H_STRIDE * 2, mbase, half, ln, acc, true);
    __syncthreads();

    // ---- GEMM2: m = silu(h1 @ mw2 + b2), K = 128 ----
    gemm_tile(hb, H_STRIDE, HID / 32, w2p, b2, mbase, half, ln, lane, acc);
    #pragma unroll
    for (int nt = 0; nt < NT; ++nt)
        #pragma unroll
        for (int v = 0; v < 8; ++v) acc[nt][v] = silu_f(acc[nt][v]);
    // segment_sum: agg[row[e]] += m  (f32 atomics into L2)
    #pragma unroll
    for (int v = 0; v < 8; ++v) {
        int er = rown[mbase + half * 8 + v];
        #pragma unroll
        for (int nt = 0; nt < NT; ++nt)
            atomicAdd(&agg[er * HID + nt * 16 + ln], acc[nt][v]);
    }
    __syncthreads();                       // everyone done reading h1
    store_bf_tile((unsigned short*)hb, H_STRIDE * 2, mbase, half, ln, acc, false);
    __syncthreads();

    // ---- GEMM3: h3 = silu(m @ cw1 + cb1), K = 128 ----
    gemm_tile(hb, H_STRIDE, HID / 32, c1p, cb1, mbase, half, ln, lane, acc);
    // ef tile is dead since GEMM1 (barriers since) -> reuse for h3
    store_bf_tile((unsigned short*)ef, EF_STRIDE * 2, mbase, half, ln, acc, true);
    __syncthreads();

    // ---- w = h3 @ cw2 + cb2 ; coords[row] += w * diff ----
    if (tid < 64) {
        const unsigned short* h3 = (const unsigned short*)ef + tid * (EF_STRIDE * 2);
        float w = cb2[0];
        #pragma unroll 4
        for (int k = 0; k < HID; ++k) w += bf2f(h3[k]) * cw2[k];
        int r = rown[tid];
        atomicAdd(&coords_out[r * 3 + 0], w * dif[tid * 3 + 0]);
        atomicAdd(&coords_out[r * 3 + 1], w * dif[tid * 3 + 1]);
        atomicAdd(&coords_out[r * 3 + 2], w * dif[tid * 3 + 2]);
    }
}

// ---------------- Node feature update ----------------
// feats' = silu([feats, agg] @ fw + fb), K = 256. One 16-node tile per wave.
__global__ __launch_bounds__(256) void node_kernel(
    const unsigned* __restrict__ feats_pairs,    // [N][64] bf16 pairs
    const float* __restrict__ agg,               // [N][128] f32
    const unsigned* __restrict__ fwp, const float* __restrict__ fb,
    unsigned short* __restrict__ feats_out) {    // [N][128] bf16
    const int tid  = threadIdx.x;
    const int lane = tid & 31;
    const int wave = tid >> 5;
    const int half = (lane >> 4) & 1;
    const int ln   = lane & 15;
    int base = (blockIdx.x * 8 + wave) * 16;
    if (base >= NNODES) return;                  // wave-uniform
    int rowc = base + ln;
    if (rowc >= NNODES) rowc = NNODES - 1;       // clamp reads; writes predicated

    v8f acc[NT];
    #pragma unroll
    for (int nt = 0; nt < NT; ++nt) {
        float b = fb[nt * 16 + ln];
        #pragma unroll
        for (int i = 0; i < 8; ++i) acc[nt][i] = b;
    }

    // K 0..127 from bf16 feats (direct fragment loads)
    const unsigned* fp = &feats_pairs[rowc * 64];
    #pragma unroll
    for (int ks = 0; ks < 4; ++ks) {
        Frag a;
        a.q[0] = *(const uint4*)(fp + ks * 16 + half * 8);
        a.q[1] = *(const uint4*)(fp + ks * 16 + half * 8 + 4);
        #pragma unroll
        for (int nt = 0; nt < NT; ++nt) {
            Frag b;
            const uint4* bp = (const uint4*)&fwp[((ks * NT + nt) * 32 + lane) * 8];
            b.q[0] = bp[0]; b.q[1] = bp[1];
            acc[nt] = wmma_bf(a, b, acc[nt]);
        }
    }
    // K 128..255 from f32 agg (convert to bf16 on the fly)
    const float* ap = &agg[rowc * HID];
    #pragma unroll
    for (int ks = 0; ks < 4; ++ks) {
        int k0 = ks * 32 + (half ? 16 : 0);
        int k1 = ks * 32 + (half ? 24 : 8);
        unsigned p[8];
        #pragma unroll
        for (int j = 0; j < 4; ++j) p[j]     = packbf(ap[k0 + 2 * j], ap[k0 + 2 * j + 1]);
        #pragma unroll
        for (int j = 0; j < 4; ++j) p[4 + j] = packbf(ap[k1 + 2 * j], ap[k1 + 2 * j + 1]);
        Frag a;
        a.q[0] = make_uint4(p[0], p[1], p[2], p[3]);
        a.q[1] = make_uint4(p[4], p[5], p[6], p[7]);
        #pragma unroll
        for (int nt = 0; nt < NT; ++nt) {
            Frag b;
            const uint4* bp = (const uint4*)&fwp[(((4 + ks) * NT + nt) * 32 + lane) * 8];
            b.q[0] = bp[0]; b.q[1] = bp[1];
            acc[nt] = wmma_bf(a, b, acc[nt]);
        }
    }
    #pragma unroll
    for (int v = 0; v < 8; ++v) {
        int row = base + half * 8 + v;
        if (row < NNODES) {
            #pragma unroll
            for (int nt = 0; nt < NT; ++nt)
                feats_out[row * HID + nt * 16 + ln] = f2bf(silu_f(acc[nt][v]));
        }
    }
}

// ---------------- Energy head (negligible FLOPs -> plain VALU) ----------------
__global__ __launch_bounds__(128) void energy_kernel(
    const unsigned short* __restrict__ feats,    // bf16 [N][128]
    const float* __restrict__ ew1, const float* __restrict__ eb1,
    const float* __restrict__ ew2, const float* __restrict__ eb2,
    float* __restrict__ out) {
    __shared__ float fv[HID];
    __shared__ float red[HID];
    int n = blockIdx.x;
    int j = threadIdx.x;
    fv[j] = bf2f(feats[n * HID + j]);
    __syncthreads();
    float t = eb1[j];
    #pragma unroll 8
    for (int k = 0; k < HID; ++k) t += fv[k] * ew1[k * HID + j];
    red[j] = silu_f(t) * ew2[j];
    __syncthreads();
    for (int s = 64; s > 0; s >>= 1) {
        if (j < s) red[j] += red[j + s];
        __syncthreads();
    }
    if (j == 0) atomicAdd(out, red[0] + eb2[0]);
}

// ---------------- Host orchestration ----------------
extern "C" void kernel_launch(void* const* d_in, const int* in_sizes, int n_in,
                              void* d_out, int out_size, void* d_ws, size_t ws_size,
                              hipStream_t stream) {
    const int*   an      = (const int*)d_in[0];
    const float* coords0 = (const float*)d_in[1];
    const int*   eidx    = (const int*)d_in[2];
    const int*   erow    = eidx;
    const int*   ecol    = eidx + NEDGES;
    const float* embed   = (const float*)d_in[3];
    const float* mw1     = (const float*)d_in[4];
    const float* mb1     = (const float*)d_in[5];
    const float* mw2     = (const float*)d_in[6];
    const float* mb2     = (const float*)d_in[7];
    const float* cw1     = (const float*)d_in[8];
    const float* cb1     = (const float*)d_in[9];
    const float* cw2     = (const float*)d_in[10];
    const float* cb2     = (const float*)d_in[11];
    const float* fw      = (const float*)d_in[12];
    const float* fb      = (const float*)d_in[13];
    const float* ew1     = (const float*)d_in[14];
    const float* eb1     = (const float*)d_in[15];
    const float* ew2     = (const float*)d_in[16];
    const float* eb2     = (const float*)d_in[17];
    (void)in_sizes; (void)n_in; (void)out_size; (void)ws_size;

    // Workspace carve (256B aligned); total ~33 MB.
    char* ws = (char*)d_ws;
    auto carve = [&](size_t bytes) -> char* {
        char* p = ws;
        ws += (bytes + 255) & ~(size_t)255;
        return p;
    };
    unsigned* featsA  = (unsigned*)carve((size_t)NNODES * 64 * 4);
    unsigned* featsB  = (unsigned*)carve((size_t)NNODES * 64 * 4);
    float*    agg     = (float*)carve((size_t)NNODES * HID * 4);
    float*    coordsA = (float*)carve((size_t)NNODES * 3 * 4);
    float*    coordsB = (float*)carve((size_t)NNODES * 3 * 4);
    unsigned* tail    = (unsigned*)carve((size_t)NEDGES * 32 * 4);
    const int W1U = (KP1 / 32) * NT * 32 * 8;     // 20480 uints / layer
    const int W2U = (HID / 32) * NT * 32 * 8;     // 8192
    const int FWU = (256 / 32) * NT * 32 * 8;     // 16384
    unsigned* w1p = (unsigned*)carve((size_t)NLAYERS * W1U * 4);
    unsigned* w2p = (unsigned*)carve((size_t)NLAYERS * W2U * 4);
    unsigned* c1p = (unsigned*)carve((size_t)NLAYERS * W2U * 4);
    unsigned* fwp = (unsigned*)carve((size_t)NLAYERS * FWU * 4);

    // --- prep ---
    init_kernel<<<(NNODES * 64 + 255) / 256, 256, 0, stream>>>(an, embed, coords0,
                                                               featsA, coordsA);
    edge_tail_kernel<<<(NEDGES + 255) / 256, 256, 0, stream>>>(erow, ecol, coords0, tail);
    for (int l = 0; l < NLAYERS; ++l) {
        pack_kernel<<<(W1U + 255) / 256, 256, 0, stream>>>(mw1 + (size_t)l * 307 * HID,
                                                           w1p + (size_t)l * W1U,
                                                           KP1 / 32, 307, 1);
        pack_kernel<<<(W2U + 255) / 256, 256, 0, stream>>>(mw2 + (size_t)l * HID * HID,
                                                           w2p + (size_t)l * W2U,
                                                           HID / 32, HID, 0);
        pack_kernel<<<(W2U + 255) / 256, 256, 0, stream>>>(cw1 + (size_t)l * HID * HID,
                                                           c1p + (size_t)l * W2U,
                                                           HID / 32, HID, 0);
        pack_kernel<<<(FWU + 255) / 256, 256, 0, stream>>>(fw + (size_t)l * 256 * HID,
                                                           fwp + (size_t)l * FWU,
                                                           256 / 32, 256, 0);
    }

    // --- 6 message-passing layers ---
    unsigned* fc = featsA;  unsigned* fn = featsB;
    float*    ca = coordsA; float*    cn = coordsB;
    for (int l = 0; l < NLAYERS; ++l) {
        layer_prep_kernel<<<(NNODES * HID + 255) / 256, 256, 0, stream>>>(agg, ca, cn);
        edge_kernel<<<NEDGES / 64, 128, 0, stream>>>(
            fc, ca, cn, agg, erow, ecol, tail,
            w1p + (size_t)l * W1U, mb1 + (size_t)l * HID,
            w2p + (size_t)l * W2U, mb2 + (size_t)l * HID,
            c1p + (size_t)l * W2U, cb1 + (size_t)l * HID,
            cw2 + (size_t)l * HID, cb2 + l);
        int ntiles = (NNODES + 15) / 16;
        node_kernel<<<(ntiles + 7) / 8, 256, 0, stream>>>(fc, agg,
                                                          fwp + (size_t)l * FWU,
                                                          fb + (size_t)l * HID,
                                                          (unsigned short*)fn);
        unsigned* tf = fc; fc = fn; fn = tf;
        float*    tc = ca; ca = cn; cn = tc;
    }

    // --- energy head ---
    zero_out_kernel<<<1, 32, 0, stream>>>((float*)d_out);
    energy_kernel<<<NNODES, 128, 0, stream>>>((const unsigned short*)fc,
                                              ew1, eb1, ew2, eb2, (float*)d_out);
}
